// EnergyDipolesMACE_60559038874220
// MI455X (gfx1250) — compile-verified, matrix operand order checked by reading, not run on hardware
//
#include <hip/hip_runtime.h>
#include <math.h>

// ---------------------------------------------------------------------------
// MACE-like energy/dipole/forces for gfx1250 (MI455X).
// GEMMs use v_wmma_f32_16x16x32_f16 with LDS weights stored fragment-major
// (ds_load_b128 fragments). Node tensors are k-major planes [k][n][c] so
// A-fragments are contiguous float4 loads and scatter atomics coalesce.
// ---------------------------------------------------------------------------

constexpr int    NNI  = 16384;
constexpr int    EEI  = 131072;
constexpr size_t NN   = 16384;
constexpr size_t EE   = 131072;
constexpr size_t NC   = NN*64;        // one [n][c] plane
constexpr int    HPLANES = 9;

typedef _Float16 v16h __attribute__((ext_vector_type(16)));
typedef _Float16 v8h  __attribute__((ext_vector_type(8)));
typedef float    v8f  __attribute__((ext_vector_type(8)));

union HFrag { v16h v; _Float16 h[16]; };
union FFrag { v8f  v; float    f[8];  };

__constant__ int d_lidx[9] = {0,1,1,1,2,2,2,2,2};

__device__ __forceinline__ float rcpf(float x){ return __builtin_amdgcn_rcpf(x); }
__device__ __forceinline__ float siluf(float x){
  float s = rcpf(1.f+__expf(-x)); return x*s;
}
__device__ __forceinline__ float dsiluf(float x){
  float s = rcpf(1.f+__expf(-x)); return s*(1.f + x*(1.f-s));
}
__device__ __forceinline__ int fragkh(int j){ return (j<8)? j : j+8; }
__device__ __forceinline__ v8f vzero(){
  v8f z = {0.f,0.f,0.f,0.f,0.f,0.f,0.f,0.f}; return z;
}
__device__ __forceinline__ v8f wmma16(const HFrag& a, const HFrag& b, v8f c){
  return __builtin_amdgcn_wmma_f32_16x16x32_f16(false, a.v, false, b.v,
                                                (short)0, c, false, false);
}
// Contiguous LDS fragment: halves [0..7] at base, [8..15] at base+16.
__device__ __forceinline__ HFrag ldsFrag(const _Float16* base){
  const v8h* p = (const v8h*)base;
  v8h lo = p[0], hi = p[2];
  HFrag r;
  r.v = __builtin_shufflevector(lo, hi, 0,1,2,3,4,5,6,7,8,9,10,11,12,13,14,15);
  return r;
}
// Strided LDS fragment (row stride between k's).
__device__ __forceinline__ HFrag ldsFragStrided(const _Float16* W, int stride,
                                                int kbase, int col){
  HFrag r;
  #pragma unroll
  for (int j=0;j<16;++j) r.h[j] = W[(kbase+fragkh(j))*stride + col];
  return r;
}
// Contiguous global f32 fragment: floats [0..7] at base, [8..15] at base+16.
__device__ __forceinline__ HFrag gmemFragF32(const float* base){
  const float4* p = (const float4*)base;
  float4 q0=p[0], q1=p[1], q2=p[4], q3=p[5];
  HFrag r;
  r.h[0]=(_Float16)q0.x; r.h[1]=(_Float16)q0.y; r.h[2]=(_Float16)q0.z; r.h[3]=(_Float16)q0.w;
  r.h[4]=(_Float16)q1.x; r.h[5]=(_Float16)q1.y; r.h[6]=(_Float16)q1.z; r.h[7]=(_Float16)q1.w;
  r.h[8]=(_Float16)q2.x; r.h[9]=(_Float16)q2.y; r.h[10]=(_Float16)q2.z; r.h[11]=(_Float16)q2.w;
  r.h[12]=(_Float16)q3.x; r.h[13]=(_Float16)q3.y; r.h[14]=(_Float16)q3.z; r.h[15]=(_Float16)q3.w;
  return r;
}
// ef fragment: K padded 8->32 with zeros (only lanes 0-15, j 0-7 carry data).
__device__ __forceinline__ HFrag efFrag(const float* efrow, int lane){
  HFrag a;
  #pragma unroll
  for (int j=0;j<16;++j) a.h[j] = (_Float16)0.f;
  if (!(lane & 16)){
    const float4* p = (const float4*)efrow;
    float4 q0=p[0], q1=p[1];
    a.h[0]=(_Float16)q0.x; a.h[1]=(_Float16)q0.y; a.h[2]=(_Float16)q0.z; a.h[3]=(_Float16)q0.w;
    a.h[4]=(_Float16)q1.x; a.h[5]=(_Float16)q1.y; a.h[6]=(_Float16)q1.z; a.h[7]=(_Float16)q1.w;
  }
  return a;
}

// ------------------------------ forward ------------------------------------

__global__ __launch_bounds__(256) void edge_geom_k(
    const float* __restrict__ pos, const float* __restrict__ shifts,
    const int* __restrict__ eidx, float* __restrict__ Y,
    float* __restrict__ ef, float* __restrict__ Lbuf)
{
  int e = blockIdx.x*256 + threadIdx.x;
  if (e >= EEI) return;
  int s = eidx[e], d = eidx[EE+e];
  float vx = pos[d*3+0]-pos[s*3+0]+shifts[(size_t)e*3+0];
  float vy = pos[d*3+1]-pos[s*3+1]+shifts[(size_t)e*3+1];
  float vz = pos[d*3+2]-pos[s*3+2]+shifts[(size_t)e*3+2];
  float L = sqrtf(vx*vx+vy*vy+vz*vz+1e-12f);
  Lbuf[e] = L;
  float ri = rcpf(L);
  float x = vx*ri, y = vy*ri, z = vz*ri;
  const float s3=1.7320508075688772f, s5=2.23606797749979f, s15=3.872983346207417f;
  float* Ye = Y + (size_t)e*9;
  Ye[0]=1.f; Ye[1]=s3*y; Ye[2]=s3*z; Ye[3]=s3*x;
  Ye[4]=s15*x*y; Ye[5]=s15*y*z; Ye[6]=0.5f*s5*(3.f*z*z-1.f);
  Ye[7]=s15*x*z; Ye[8]=0.5f*s15*(x*x-y*y);
  float u = L*0.2f;
  float env = 0.f;
  if (u < 1.f){
    float u5 = u*u*u*u*u;
    env = 1.f - 21.f*u5 + 35.f*u5*u - 15.f*u5*u*u;
  }
  const float PI_ = 3.14159265358979f, sq = 0.6324555320336759f; // sqrt(2/5)
  float se = sq*ri*env;
  for (int n=1;n<=8;++n)
    ef[(size_t)e*8 + n-1] = se*__sinf((float)n*PI_*u);
}

__global__ __launch_bounds__(256) void h0_init_k(
    const float* __restrict__ attrs, const float* __restrict__ Wemb,
    float* __restrict__ h0)
{
  int idx = blockIdx.x*256 + threadIdx.x;
  if (idx >= NNI*64) return;
  int n = idx>>6, c = idx&63;
  float v = 0.f;
  for (int el=0; el<10; ++el) v += attrs[(size_t)n*10+el]*Wemb[el*64+c];
  h0[idx] = v;
  for (int k=1;k<9;++k) h0[(size_t)k*NC + idx] = 0.f;
}

__global__ __launch_bounds__(256) void e0_k(
    const float* __restrict__ attrs, const float* __restrict__ aE,
    const int* __restrict__ batch, float* __restrict__ energy)
{
  int n = blockIdx.x*256 + threadIdx.x;
  if (n >= NNI) return;
  float v = 0.f;
  for (int el=0; el<10; ++el) v += attrs[(size_t)n*10+el]*aE[el];
  atomicAdd(&energy[batch[n]], v);
}

// Radial MLP forward: ef(E x 8) -> silu chain -> R(E x 192, [e][l*64+c] order).
__global__ __launch_bounds__(128) void radial_fwd_k(
    const float* __restrict__ ef, const float* __restrict__ W1,
    const float* __restrict__ W2, const float* __restrict__ W3,
    const float* __restrict__ Wtp, float* __restrict__ R)
{
  __shared__ _Float16 sW1cm[64*32];     // [col][k] K padded 8->32
  __shared__ _Float16 sW2cm[64*64];     // [col][k]
  __shared__ _Float16 sW3cm[64*64];
  __shared__ _Float16 sWtpCm[192*64];   // [col'][k], col' = l*64+c permutation
  __shared__ _Float16 act[4][16*64];
  int t = threadIdx.x;
  for (int i=t;i<64*32;i+=128){
    int col=i>>5, k=i&31;
    sW1cm[i] = (_Float16)((k<8)? W1[k*64+col] : 0.f);
  }
  for (int i=t;i<64*64;i+=128){
    int col=i>>6, k=i&63;
    sW2cm[i] = (_Float16)W2[k*64+col];
    sW3cm[i] = (_Float16)W3[k*64+col];
  }
  for (int i=t;i<192*64;i+=128){
    int col=i>>6, k=i&63;
    int orig = (col&63)*3 + (col>>6);           // [e][c*3+l] -> [e][l*64+c]
    sWtpCm[i] = (_Float16)Wtp[k*192 + orig];
  }
  __syncthreads();

  int wave = t>>5, lane = t&31;
  int row = lane&15, koff = (lane&16)?8:0, rbase = (lane&16)?8:0;
  size_t e0 = (size_t)blockIdx.x*64 + wave*16;
  _Float16* A = act[wave];

  // z1 = ef_pad @ W1
  HFrag a = efFrag(ef + (e0+row)*8, lane);
  FFrag z[4];
  for (int ct=0;ct<4;++ct){
    HFrag b = ldsFrag(&sW1cm[(ct*16+row)*32 + koff]);
    z[ct].v = wmma16(a,b,vzero());
  }
  for (int ct=0;ct<4;++ct)
    #pragma unroll
    for (int v=0;v<8;++v) A[(rbase+v)*64 + ct*16+row] = (_Float16)siluf(z[ct].f[v]);
  __syncthreads();

  // z2 = r1 @ W2 ; z3 = r2 @ W3
  const _Float16* Ws[2] = {sW2cm, sW3cm};
  for (int layer=0; layer<2; ++layer){
    for (int ct=0;ct<4;++ct) z[ct].v = vzero();
    for (int kc=0;kc<64;kc+=32){
      HFrag aa = ldsFrag(&A[row*64 + kc + koff]);
      for (int ct=0;ct<4;++ct){
        HFrag b = ldsFrag(&Ws[layer][(ct*16+row)*64 + kc + koff]);
        z[ct].v = wmma16(aa,b,z[ct].v);
      }
    }
    __syncthreads();
    for (int ct=0;ct<4;++ct)
      #pragma unroll
      for (int v=0;v<8;++v) A[(rbase+v)*64 + ct*16+row] = (_Float16)siluf(z[ct].f[v]);
    __syncthreads();
  }

  // R = r3 @ Wtp (12 col tiles, permuted columns)
  FFrag rAcc[12];
  for (int ct=0;ct<12;++ct) rAcc[ct].v = vzero();
  for (int kc=0;kc<64;kc+=32){
    HFrag aa = ldsFrag(&A[row*64 + kc + koff]);
    for (int ct=0;ct<12;++ct){
      HFrag b = ldsFrag(&sWtpCm[(ct*16+row)*64 + kc + koff]);
      rAcc[ct].v = wmma16(aa,b,rAcc[ct].v);
    }
  }
  for (int ct=0;ct<12;++ct){
    int col = ct*16+row;
    #pragma unroll
    for (int v=0;v<8;++v) R[(e0+rbase+v)*192 + col] = rAcc[ct].f[v];
  }
}

// Message + scatter: agg[k][dst][c] += h[0][src][c]*Y[e,k]*R[e][lidx[k]*64+c]/16
__global__ __launch_bounds__(256) void message_fwd_k(
    const float* __restrict__ hin, const float* __restrict__ Y,
    const float* __restrict__ R, const int* __restrict__ eidx,
    float* __restrict__ agg)
{
  int t = threadIdx.x; int eL = t>>6, c = t&63;
  size_t e = (size_t)blockIdx.x*4 + eL;
  int s = eidx[e], d = eidx[EE+e];
  __builtin_prefetch(&R[(e+4)*192 + c], 0, 0);
  float hs = hin[(size_t)s*64 + c];
  float Rv[3]; for (int l=0;l<3;++l) Rv[l] = R[e*192 + l*64 + c];
  float Yv[9]; for (int k=0;k<9;++k) Yv[k] = Y[e*9+k];
  for (int k=0;k<9;++k)
    atomicAdd(&agg[(size_t)k*NC + (size_t)d*64 + c],
              hs*Yv[k]*Rv[d_lidx[k]]*(1.f/16.f));
}

// Generic per-k node GEMM on planes: out[k][n][:] = in[k][n][:] @ B(lidx[k])
__global__ __launch_bounds__(256) void node_gemm_k(
    const float* __restrict__ in, const float* __restrict__ Bsrc,
    int nB, int transB, const float* __restrict__ bias,
    float* __restrict__ out)
{
  __shared__ _Float16 Bh[3*64*64];       // [l][col][k] fragment-major
  int t = threadIdx.x;
  int total = nB*4096;
  for (int i=t;i<total;i+=256){
    int l=i>>12, rc=i&4095, col=rc>>6, k=rc&63;
    float v = transB ? Bsrc[l*4096 + rc] : Bsrc[l*4096 + k*64 + col];
    Bh[i] = (_Float16)v;
  }
  __syncthreads();
  int wave=t>>5, lane=t&31;
  int row=lane&15, koff=(lane&16)?8:0, rbase=(lane&16)?8:0;
  size_t n0 = (size_t)blockIdx.x*16;
  for (int kk=wave; kk<9; kk+=8){
    const _Float16* B = Bh + ((nB==1)?0:d_lidx[kk])*4096;
    FFrag acc[4];
    for (int ct=0;ct<4;++ct) acc[ct].v = vzero();
    for (int kc=0;kc<64;kc+=32){
      HFrag a = gmemFragF32(in + (size_t)kk*NC + (n0+row)*64 + kc + koff);
      for (int ct=0;ct<4;++ct){
        HFrag b = ldsFrag(&B[(ct*16+row)*64 + kc + koff]);
        acc[ct].v = wmma16(a,b,acc[ct].v);
      }
    }
    for (int ct=0;ct<4;++ct){
      int col = ct*16+row;
      #pragma unroll
      for (int v=0;v<8;++v){
        float val = acc[ct].f[v];
        if (kk==0 && bias) val += bias[col];
        out[(size_t)kk*NC + (n0+rbase+v)*64 + col] = val;
      }
    }
  }
}

// h_new = mixed*(1+Wprod*mixed0)+sc ; energy & per-node dipole readouts.
__global__ __launch_bounds__(64) void combine_k(
    const float* __restrict__ mixed, const float* __restrict__ sc,
    const float* __restrict__ Wprod, const float* __restrict__ we,
    const float* __restrict__ wd, const int* __restrict__ batch,
    float* __restrict__ hout, float* __restrict__ energy, float* __restrict__ ndip)
{
  __shared__ float red[64*4];
  size_t n = blockIdx.x; int d = threadIdx.x;
  size_t base = n*64 + d;
  float wp = Wprod[d];
  float m0 = mixed[base];
  float fac = 1.f + wp*m0;
  float eloc=0.f, dx=0.f, dy=0.f, dz=0.f;
  for (int k=0;k<9;++k){
    float h = mixed[(size_t)k*NC + base]*fac + sc[(size_t)k*NC + base];
    hout[(size_t)k*NC + base] = h;
    if      (k==0) eloc = h*we[d];
    else if (k==1) dx = h*wd[d];
    else if (k==2) dy = h*wd[d];
    else if (k==3) dz = h*wd[d];
  }
  red[d*4+0]=eloc; red[d*4+1]=dx; red[d*4+2]=dy; red[d*4+3]=dz;
  __syncthreads();
  for (int s=32;s>0;s>>=1){
    if (d<s) for (int q=0;q<4;++q) red[d*4+q]+=red[(d+s)*4+q];
    __syncthreads();
  }
  if (d==0){
    atomicAdd(&energy[batch[n]], red[0]);
    ndip[n*3+0]+=red[1]; ndip[n*3+1]+=red[2]; ndip[n*3+2]+=red[3];
  }
}

// ------------------------------ backward -----------------------------------

__global__ __launch_bounds__(256) void init_gh_k(
    const float* __restrict__ weLast, float* __restrict__ gh)
{
  int idx = blockIdx.x*256 + threadIdx.x;
  if (idx >= NNI*64) return;
  int d = idx&63;
  gh[idx] = weLast[d];
  for (int k=1;k<9;++k) gh[(size_t)k*NC + idx] = 0.f;
}

__global__ __launch_bounds__(64) void gmixed_k(
    const float* __restrict__ gh, const float* __restrict__ mixed,
    const float* __restrict__ Wprod, float* __restrict__ gmix)
{
  size_t n = blockIdx.x; int d = threadIdx.x;
  size_t base = n*64 + d;
  float wp = Wprod[d];
  float m0 = mixed[base];
  float fac = 1.f + wp*m0;
  float s = 0.f;
  float g[9];
  for (int k=0;k<9;++k){
    float gv = gh[(size_t)k*NC + base];
    s += gv*mixed[(size_t)k*NC + base];
    g[k] = gv*fac;
  }
  g[0] += wp*s;
  for (int k=0;k<9;++k) gmix[(size_t)k*NC + base] = g[k];
}

// Edge backward: g_h_src0 atomics, gR (in-place over R), gY.
__global__ __launch_bounds__(256) void edge_bwd_k(
    const float* __restrict__ gagg, const float* __restrict__ hin,
    const float* __restrict__ Y, float* __restrict__ Rbuf,
    const int* __restrict__ eidx, float* __restrict__ ghprev,
    float* __restrict__ gY)
{
  __shared__ float sgY[4][9];
  int t = threadIdx.x; int eL = t>>6, c = t&63;
  size_t e = (size_t)blockIdx.x*4 + eL;
  if (t < 36) sgY[t/9][t%9] = 0.f;
  __syncthreads();
  int s = eidx[e], d = eidx[EE+e];
  float hs = hin[(size_t)s*64 + c];
  float Yv[9]; for (int k=0;k<9;++k) Yv[k] = Y[e*9+k];
  float Rv[3]; for (int l=0;l<3;++l) Rv[l] = Rbuf[e*192 + l*64 + c];
  float ga[9]; for (int k=0;k<9;++k)
    ga[k] = gagg[(size_t)k*NC + (size_t)d*64 + c]*(1.f/16.f);
  float gh0 = 0.f; float gR[3] = {0.f,0.f,0.f};
  for (int k=0;k<9;++k){
    int l = d_lidx[k];
    float t1 = ga[k]*Yv[k];
    gh0   += t1*Rv[l];
    gR[l] += t1*hs;
    atomicAdd(&sgY[eL][k], ga[k]*hs*Rv[l]);
  }
  atomicAdd(&ghprev[(size_t)s*64 + c], gh0);
  for (int l=0;l<3;++l) Rbuf[e*192 + l*64 + c] = gR[l];
  __syncthreads();
  if (c < 9) gY[e*9 + c] = sgY[eL][c];
}

// Radial MLP backward with in-kernel forward recompute (z kept in VGPRs).
__global__ __launch_bounds__(128) void radial_bwd_k(
    const float* __restrict__ ef, const float* __restrict__ gR,
    const float* __restrict__ W1, const float* __restrict__ W2,
    const float* __restrict__ W3, const float* __restrict__ Wtp,
    float* __restrict__ gef)
{
  __shared__ _Float16 sW1rm[32*64];     // padded rows >=8 are zero
  __shared__ _Float16 sW2rm[64*64];
  __shared__ _Float16 sW3rm[64*64];
  __shared__ _Float16 sWtpRm[64*192];   // [c][j'], j' permuted as l*64+c
  __shared__ _Float16 act[4][16*64];
  int t = threadIdx.x;
  for (int i=t;i<32*64;i+=128)  sW1rm[i] = (_Float16)((i<8*64)? W1[i] : 0.f);
  for (int i=t;i<64*64;i+=128){ sW2rm[i] = (_Float16)W2[i]; sW3rm[i] = (_Float16)W3[i]; }
  for (int i=t;i<64*192;i+=128){
    int c=i/192, jp=i%192;
    int orig = (jp&63)*3 + (jp>>6);
    sWtpRm[i] = (_Float16)Wtp[c*192 + orig];
  }
  __syncthreads();

  int wave = t>>5, lane = t&31;
  int row = lane&15, koff = (lane&16)?8:0, rbase = (lane&16)?8:0;
  size_t e0 = (size_t)blockIdx.x*64 + wave*16;
  _Float16* A = act[wave];

  // ---- forward recompute: z1, z2, z3 stay in registers ----
  HFrag a = efFrag(ef + (e0+row)*8, lane);
  FFrag z1[4], z2[4], z3[4];
  for (int ct=0;ct<4;++ct){
    HFrag b = ldsFragStrided(sW1rm, 64, koff, ct*16+row);
    z1[ct].v = wmma16(a,b,vzero());
  }
  for (int ct=0;ct<4;++ct)
    #pragma unroll
    for (int v=0;v<8;++v) A[(rbase+v)*64 + ct*16+row] = (_Float16)siluf(z1[ct].f[v]);
  __syncthreads();
  for (int ct=0;ct<4;++ct) z2[ct].v = vzero();
  for (int kc=0;kc<64;kc+=32){
    HFrag aa = ldsFrag(&A[row*64 + kc + koff]);
    for (int ct=0;ct<4;++ct){
      HFrag b = ldsFragStrided(sW2rm, 64, kc+koff, ct*16+row);
      z2[ct].v = wmma16(aa,b,z2[ct].v);
    }
  }
  __syncthreads();
  for (int ct=0;ct<4;++ct)
    #pragma unroll
    for (int v=0;v<8;++v) A[(rbase+v)*64 + ct*16+row] = (_Float16)siluf(z2[ct].f[v]);
  __syncthreads();
  for (int ct=0;ct<4;++ct) z3[ct].v = vzero();
  for (int kc=0;kc<64;kc+=32){
    HFrag aa = ldsFrag(&A[row*64 + kc + koff]);
    for (int ct=0;ct<4;++ct){
      HFrag b = ldsFragStrided(sW3rm, 64, kc+koff, ct*16+row);
      z3[ct].v = wmma16(aa,b,z3[ct].v);
    }
  }
  __syncthreads();

  // ---- backward: g_r3 = gR @ Wtp^T (K=192, permuted order) ----
  FFrag g[4];
  for (int ct=0;ct<4;++ct) g[ct].v = vzero();
  for (int kc=0;kc<192;kc+=32){
    HFrag ag = gmemFragF32(gR + (e0+row)*192 + kc + koff);
    for (int ct=0;ct<4;++ct){
      HFrag b = ldsFrag(&sWtpRm[(ct*16+row)*192 + kc + koff]);
      g[ct].v = wmma16(ag,b,g[ct].v);
    }
  }
  for (int ct=0;ct<4;++ct)
    #pragma unroll
    for (int v=0;v<8;++v)
      A[(rbase+v)*64 + ct*16+row] = (_Float16)(g[ct].f[v]*dsiluf(z3[ct].f[v]));
  __syncthreads();
  // g_r2 = g_z3 @ W3^T  (contiguous rows of row-major W3)
  for (int ct=0;ct<4;++ct) g[ct].v = vzero();
  for (int kc=0;kc<64;kc+=32){
    HFrag aa = ldsFrag(&A[row*64 + kc + koff]);
    for (int ct=0;ct<4;++ct){
      HFrag b = ldsFrag(&sW3rm[(ct*16+row)*64 + kc + koff]);
      g[ct].v = wmma16(aa,b,g[ct].v);
    }
  }
  __syncthreads();
  for (int ct=0;ct<4;++ct)
    #pragma unroll
    for (int v=0;v<8;++v)
      A[(rbase+v)*64 + ct*16+row] = (_Float16)(g[ct].f[v]*dsiluf(z2[ct].f[v]));
  __syncthreads();
  // g_r1 = g_z2 @ W2^T
  for (int ct=0;ct<4;++ct) g[ct].v = vzero();
  for (int kc=0;kc<64;kc+=32){
    HFrag aa = ldsFrag(&A[row*64 + kc + koff]);
    for (int ct=0;ct<4;++ct){
      HFrag b = ldsFrag(&sW2rm[(ct*16+row)*64 + kc + koff]);
      g[ct].v = wmma16(aa,b,g[ct].v);
    }
  }
  __syncthreads();
  for (int ct=0;ct<4;++ct)
    #pragma unroll
    for (int v=0;v<8;++v)
      A[(rbase+v)*64 + ct*16+row] = (_Float16)(g[ct].f[v]*dsiluf(z1[ct].f[v]));
  __syncthreads();
  // g_ef = g_z1 @ W1^T (one 16-col tile; padded rows of sW1rm are zero)
  FFrag ge; ge.v = vzero();
  for (int kc=0;kc<64;kc+=32){
    HFrag aa = ldsFrag(&A[row*64 + kc + koff]);
    HFrag b  = ldsFrag(&sW1rm[row*64 + kc + koff]);
    ge.v = wmma16(aa,b,ge.v);
  }
  if (row < 8)
    #pragma unroll
    for (int v=0;v<8;++v) gef[(e0+rbase+v)*8 + row] = ge.f[v];
}

// Geometric backward: gY,gef -> g_vec -> force atomics.
__global__ __launch_bounds__(256) void geom_bwd_k(
    const float* __restrict__ pos, const float* __restrict__ shifts,
    const int* __restrict__ eidx, const float* __restrict__ Lbuf,
    const float* __restrict__ gef, const float* __restrict__ gY,
    float* __restrict__ forces)
{
  int e = blockIdx.x*256 + threadIdx.x;
  if (e >= EEI) return;
  int s = eidx[e], d = eidx[EE+e];
  float vx = pos[d*3+0]-pos[s*3+0]+shifts[(size_t)e*3+0];
  float vy = pos[d*3+1]-pos[s*3+1]+shifts[(size_t)e*3+1];
  float vz = pos[d*3+2]-pos[s*3+2]+shifts[(size_t)e*3+2];
  float L = Lbuf[e];
  float ri = rcpf(L);
  float x = vx*ri, y = vy*ri, z = vz*ri;
  // g_L through radial embedding
  float u = L*0.2f;
  float env = 0.f, denvdu = 0.f;
  if (u < 1.f){
    float u4 = u*u*u*u; float u5 = u4*u; float u6 = u5*u;
    env    = 1.f - 21.f*u5 + 35.f*u6 - 15.f*u6*u;
    denvdu = -105.f*u4 + 210.f*u5 - 105.f*u6;
  }
  const float PI_ = 3.14159265358979f, sq = 0.6324555320336759f;
  float gl = 0.f;
  for (int n=1;n<=8;++n){
    float arg = (float)n*PI_*u;
    float sn = __sinf(arg), cn = __cosf(arg);
    float bess  = sq*sn*ri;
    float dbess = sq*(((float)n*PI_*0.2f)*cn*ri - sn*ri*ri);
    float defdl = dbess*env + bess*denvdu*0.2f;
    gl += gef[(size_t)e*8 + n-1]*defdl;
  }
  // g_u through spherical harmonics
  const float s3=1.7320508075688772f, s5=2.23606797749979f, s15=3.872983346207417f;
  const float* gy = gY + (size_t)e*9;
  float g1=gy[1],g2=gy[2],g3=gy[3],g4=gy[4],g5=gy[5],g6=gy[6],g7=gy[7],g8=gy[8];
  float gux = s3*g3 + s15*(y*g4 + z*g7 + x*g8);
  float guy = s3*g1 + s15*(x*g4 + z*g5 - y*g8);
  float guz = s3*g2 + s15*(y*g5 + x*g7) + 3.f*s5*z*g6;
  float dot = x*gux + y*guy + z*guz;
  float gvx = gl*x + (gux - x*dot)*ri;
  float gvy = gl*y + (guy - y*dot)*ri;
  float gvz = gl*z + (guz - z*dot)*ri;
  // forces = -dE/dpos : dst gets -g_vec, src gets +g_vec
  atomicAdd(&forces[s*3+0],  gvx);
  atomicAdd(&forces[s*3+1],  gvy);
  atomicAdd(&forces[s*3+2],  gvz);
  atomicAdd(&forces[d*3+0], -gvx);
  atomicAdd(&forces[d*3+1], -gvy);
  atomicAdd(&forces[d*3+2], -gvz);
}

__global__ __launch_bounds__(256) void dipole_k(
    const float* __restrict__ ndip, const float* __restrict__ charges,
    const float* __restrict__ pos, const int* __restrict__ batch,
    float* __restrict__ dip)
{
  int n = blockIdx.x*256 + threadIdx.x;
  if (n >= NNI) return;
  int g = batch[n];
  float q = charges[n];
  for (int j=0;j<3;++j)
    atomicAdd(&dip[g*3+j], ndip[(size_t)n*3+j] + q*pos[(size_t)n*3+j]);
}

// ------------------------------ launcher -----------------------------------

constexpr size_t HSZ = 9*NC;           // one node-feature tensor (9 planes)
constexpr size_t O_Y    = 0;
constexpr size_t O_EF   = O_Y   + EE*9;
constexpr size_t O_L    = O_EF  + EE*8;
constexpr size_t O_H0   = O_L   + EE;
constexpr size_t O_H1   = O_H0  + HSZ;
constexpr size_t O_H2   = O_H1  + HSZ;
constexpr size_t O_R0   = O_H2  + HSZ;
constexpr size_t O_R1   = O_R0  + EE*192;
constexpr size_t O_AGG  = O_R1  + EE*192;   // fwd agg / bwd g_agg
constexpr size_t O_MIX0 = O_AGG + HSZ;
constexpr size_t O_MIX1 = O_MIX0+ HSZ;
constexpr size_t O_SC   = O_MIX1+ HSZ;      // fwd sc / bwd g_mixed
constexpr size_t O_GHA  = O_SC  + HSZ;
constexpr size_t O_GHB  = O_GHA + HSZ;
constexpr size_t O_GY   = O_GHB + HSZ;
constexpr size_t O_GEF  = O_GY  + EE*9;
constexpr size_t O_NDIP = O_GEF + EE*8;

extern "C" void kernel_launch(void* const* d_in, const int* in_sizes, int n_in,
                              void* d_out, int out_size, void* d_ws, size_t ws_size,
                              hipStream_t stream) {
  (void)in_sizes; (void)n_in; (void)ws_size;
  const float* positions = (const float*)d_in[0];
  const float* node_attrs= (const float*)d_in[1];
  const float* charges   = (const float*)d_in[2];
  const float* shifts    = (const float*)d_in[3];
  const int*   eidx      = (const int*)  d_in[4];
  const int*   batch     = (const int*)  d_in[5];
  const float* W_embed   = (const float*)d_in[6];
  const float* aE        = (const float*)d_in[7];
  const float* W1s       = (const float*)d_in[8];   // [2][8][64]
  const float* W2s       = (const float*)d_in[9];   // [2][64][64]
  const float* W3s       = (const float*)d_in[10];  // [2][64][64]
  const float* Wtps      = (const float*)d_in[11];  // [2][64][192]
  const float* Wmixs     = (const float*)d_in[12];  // [2][3][64][64]
  const float* Wscs      = (const float*)d_in[13];  // [2][64][64]
  const float* Wprods    = (const float*)d_in[14];  // [2][64]
  const float* wes       = (const float*)d_in[15];  // [2][64]
  const float* wds       = (const float*)d_in[16];  // [2][64]

  float* out = (float*)d_out;        // [16 energy | 48 dipole | N*3 forces]
  float* ws  = (float*)d_ws;
  float* Y    = ws + O_Y;   float* ef   = ws + O_EF;  float* Lbuf = ws + O_L;
  float* hb[3]  = { ws + O_H0, ws + O_H1, ws + O_H2 };
  float* Rb[2]  = { ws + O_R0, ws + O_R1 };
  float* agg    = ws + O_AGG;
  float* mixb[2]= { ws + O_MIX0, ws + O_MIX1 };
  float* scbuf  = ws + O_SC;
  float* gha    = ws + O_GHA; float* ghb = ws + O_GHB;
  float* gYb    = ws + O_GY;  float* gefb = ws + O_GEF;
  float* ndip   = ws + O_NDIP;

  hipMemsetAsync(d_out, 0, (size_t)out_size*sizeof(float), stream);
  hipMemsetAsync(ndip, 0, NN*3*sizeof(float), stream);

  edge_geom_k<<<(EEI+255)/256, 256, 0, stream>>>(positions, shifts, eidx, Y, ef, Lbuf);
  h0_init_k<<<(NNI*64+255)/256, 256, 0, stream>>>(node_attrs, W_embed, hb[0]);
  e0_k<<<(NNI+255)/256, 256, 0, stream>>>(node_attrs, aE, batch, out);

  // -------- forward interaction layers --------
  for (int i=0;i<2;++i){
    radial_fwd_k<<<EEI/64, 128, 0, stream>>>(ef, W1s + (size_t)i*512, W2s + (size_t)i*4096,
                                             W3s + (size_t)i*4096, Wtps + (size_t)i*12288, Rb[i]);
    hipMemsetAsync(agg, 0, HSZ*sizeof(float), stream);
    message_fwd_k<<<EEI/4, 256, 0, stream>>>(hb[i], Y, Rb[i], eidx, agg);
    node_gemm_k<<<NNI/16, 256, 0, stream>>>(agg,  Wmixs + (size_t)i*3*4096, 3, 0, nullptr, mixb[i]);
    node_gemm_k<<<NNI/16, 256, 0, stream>>>(hb[i], Wscs + (size_t)i*4096,   1, 0, nullptr, scbuf);
    combine_k<<<NNI, 64, 0, stream>>>(mixb[i], scbuf, Wprods + (size_t)i*64,
                                      wes + (size_t)i*64, wds + (size_t)i*64,
                                      batch, hb[i+1], out, ndip);
  }

  // -------- backward (forces) --------
  init_gh_k<<<(NNI*64+255)/256, 256, 0, stream>>>(wes + 64, gha);
  float* gh_cur = gha; float* gh_prev = ghb;
  for (int i=1;i>=0;--i){
    gmixed_k<<<NNI, 64, 0, stream>>>(gh_cur, mixb[i], Wprods + (size_t)i*64, scbuf);
    node_gemm_k<<<NNI/16, 256, 0, stream>>>(scbuf,  Wmixs + (size_t)i*3*4096, 3, 1, nullptr, agg);
    node_gemm_k<<<NNI/16, 256, 0, stream>>>(gh_cur, Wscs + (size_t)i*4096,    1, 1,
                                            (i>0)? wes + (size_t)(i-1)*64 : nullptr, gh_prev);
    edge_bwd_k<<<EEI/4, 256, 0, stream>>>(agg, hb[i], Y, Rb[i], eidx, gh_prev, gYb);
    radial_bwd_k<<<EEI/64, 128, 0, stream>>>(ef, Rb[i], W1s + (size_t)i*512,
                                             W2s + (size_t)i*4096, W3s + (size_t)i*4096,
                                             Wtps + (size_t)i*12288, gefb);
    geom_bwd_k<<<(EEI+255)/256, 256, 0, stream>>>(positions, shifts, eidx, Lbuf,
                                                  gefb, gYb, out + 64);
    float* tmp = gh_cur; gh_cur = gh_prev; gh_prev = tmp;
  }

  dipole_k<<<(NNI+255)/256, 256, 0, stream>>>(ndip, charges, positions, batch, out + 16);
}
